// ResblockDownOneModulateBEST_44392781971494
// MI455X (gfx1250) — matrix operand order repaired and checked by reading, not compiled, not run on hardware
//
#include <hip/hip_runtime.h>

typedef unsigned short u16;
typedef __attribute__((ext_vector_type(16))) __bf16 v16bf;
typedef __attribute__((ext_vector_type(8)))  float  v8f;

union Frag { v16bf v; uint4 u[2]; };

__device__ __forceinline__ u16 f2bf(float f) {
    unsigned u = __float_as_uint(f);
    u += 0x7FFFu + ((u >> 16) & 1u);   // round-to-nearest-even
    return (u16)(u >> 16);
}
__device__ __forceinline__ float swishf(float v) { return v / (1.f + __expf(-v)); }

// Padded activation dims (pad=1 halo in D,H,W)
#define DP 10
#define HP 130
#define WP 130

// ---------------------------------------------------------------------------
// CDNA5 async global->LDS staging (ASYNCcnt-tracked DMA), with sync fallback
// Builtin signature (from hipcc diagnostic): first param is
//   'int __vector_size__(16) __device__ *'  i.e. int4 in addrspace(1).
// ---------------------------------------------------------------------------
#if defined(__has_builtin)
# if __has_builtin(__builtin_amdgcn_global_load_async_to_lds_b128) && \
     __has_builtin(__builtin_amdgcn_s_wait_asynccnt)
#  define ASYNC_LDS 1
# endif
#endif
#ifndef ASYNC_LDS
# define ASYNC_LDS 0
#endif

typedef int v4i __attribute__((__vector_size__(16)));
typedef __attribute__((address_space(1))) v4i g_v4i;   // global (AS1) int4
typedef __attribute__((address_space(3))) v4i s_v4i;   // LDS (AS3) int4

__device__ __forceinline__ void cp16_lds(u16* dst, const u16* src) {
#if ASYNC_LDS
    __builtin_amdgcn_global_load_async_to_lds_b128((g_v4i*)src, (s_v4i*)dst, 0, 0);
#else
    *(uint4*)dst = *(const uint4*)src;
#endif
}

__device__ __forceinline__ void wait_stage() {
#if ASYNC_LDS
    __builtin_amdgcn_s_wait_asynccnt(0);
#endif
    __syncthreads();
}

// ---------------------------------------------------------------------------
// One K=64 stage: A tile [rows][64], B tile [128 n][64]; 16 WMMAs per call.
// B fragments double-buffered in registers so ds_loads overlap WMMA issue.
// ---------------------------------------------------------------------------
__device__ __forceinline__ void compute_stage(const u16* cA, const u16* cB,
                                              int arow, int lane, v8f* acc)
{
    const int kbA = (lane < 16) ? 0 : 8;
    const int kbB = (lane < 16) ? 0 : 8;
    const int nbase = lane & 15;
#pragma unroll
    for (int kc = 0; kc < 2; ++kc) {
        Frag af;
        const u16* ap = cA + arow * 64 + kc * 32;
        af.u[0] = *(const uint4*)(ap + kbA);
        af.u[1] = *(const uint4*)(ap + kbA + 16);
        Frag bf[2];
        const u16* bp0 = cB + nbase * 64 + kc * 32;
        bf[0].u[0] = *(const uint4*)(bp0 + kbB);
        bf[0].u[1] = *(const uint4*)(bp0 + kbB + 16);
#pragma unroll
        for (int nb = 0; nb < 8; ++nb) {
            if (nb < 7) {       // prefetch next B fragment while WMMA nb runs
                const u16* bp = cB + ((nb + 1) * 16 + nbase) * 64 + kc * 32;
                bf[(nb + 1) & 1].u[0] = *(const uint4*)(bp + kbB);
                bf[(nb + 1) & 1].u[1] = *(const uint4*)(bp + kbB + 16);
            }
            acc[nb] = __builtin_amdgcn_wmma_f32_16x16x32_bf16(
                false, af.v, false, bf[nb & 1].v, (short)0, acc[nb], false, false);
        }
    }
}

// ---------------------------------------------------------------------------
// s1 = embed @ d1_w + d1_b  (4x64);  s2 = context @ d2_w + d2_b  (4x128)
// ---------------------------------------------------------------------------
__global__ __launch_bounds__(256) void modulate_kernel(
    const float* __restrict__ embed, const float* __restrict__ context,
    const float* __restrict__ d1w, const float* __restrict__ d1b,
    const float* __restrict__ d2w, const float* __restrict__ d2b,
    float* __restrict__ s1, float* __restrict__ s2)
{
    int idx = blockIdx.x * 256 + threadIdx.x;
    if (blockIdx.x == 0) {
        int b = idx >> 6, c = idx & 63;
        float acc = d1b[c];
        for (int e = 0; e < 256; ++e) acc += embed[b * 256 + e] * d1w[e * 64 + c];
        s1[idx] = acc;
    } else {
        int i = idx - 256;
        int b = i >> 7, n = i & 127;
        float acc = d2b[n];
        for (int e = 0; e < 256; ++e) acc += context[b * 256 + e] * d2w[e * 128 + n];
        s2[i] = acc;
    }
}

// ---------------------------------------------------------------------------
// Weight repack fp32 -> bf16, layout [tap][n][c]
// ---------------------------------------------------------------------------
__global__ __launch_bounds__(256) void prep_weights_kernel(
    const float* __restrict__ w20, const float* __restrict__ w21,
    const float* __restrict__ dwn, const float* __restrict__ resw,
    u16* __restrict__ wT20, u16* __restrict__ wT21,
    u16* __restrict__ wTdn, u16* __restrict__ resT)
{
    int idx = blockIdx.x * 256 + threadIdx.x;
    if (idx < 221184) {                                   // 27*128*64
        int t = idx / 8192, r = idx % 8192, n = r >> 6, c = r & 63;
        wT20[idx] = f2bf(w20[(size_t)n * 1728 + c * 27 + t]);
    } else if (idx < 221184 + 442368) {                   // 27*128*128
        int i = idx - 221184;
        int t = i / 16384, r = i % 16384, n = r >> 7, c = r & 127;
        wT21[i] = f2bf(w21[(size_t)n * 3456 + c * 27 + t]);
    } else if (idx < 221184 + 2 * 442368) {
        int i = idx - 221184 - 442368;
        int t = i / 16384, r = i % 16384, n = r >> 7, c = r & 127;
        wTdn[i] = f2bf(dwn[(size_t)n * 3456 + c * 27 + t]);
    } else if (idx < 221184 + 2 * 442368 + 8192) {        // res_w flat [n][c]
        int i = idx - 221184 - 2 * 442368;
        resT[i] = f2bf(resw[i]);
    }
}

// ---------------------------------------------------------------------------
__global__ __launch_bounds__(256) void zero_fill_kernel(uint4* __restrict__ p, long n)
{
    long i = (long)blockIdx.x * 256 + threadIdx.x;
    long stride = (long)gridDim.x * 256;
    uint4 zz; zz.x = zz.y = zz.z = zz.w = 0u;
    for (; i < n; i += stride) p[i] = zz;
}

// ---------------------------------------------------------------------------
// a0[b][zp][yp][xp][c] = bf16( swish(x + s1) ), padded channel-last
// ---------------------------------------------------------------------------
__global__ __launch_bounds__(256) void prep_a0_kernel(
    const float* __restrict__ x, const float* __restrict__ s1, u16* __restrict__ a0)
{
    __shared__ float tile[64][129];
    const int tid = threadIdx.x;
    const int bid = blockIdx.x;
    const int b = bid >> 10, z = (bid >> 7) & 7, y = bid & 127;

    for (int c0 = 0; c0 < 64; c0 += 2) {
        int c = c0 + (tid >> 7), w = tid & 127;
        tile[c][w] = x[(((size_t)(b * 64 + c) * 8 + z) * 128 + y) * 128 + w];
    }
    __syncthreads();
    const size_t base = ((size_t)((b * DP + z + 1) * HP + (y + 1)) * WP + 1) * 64;
    for (int i = 0; i < 32; ++i) {
        int idx = i * 256 + tid, w = idx >> 6, c = idx & 63;
        a0[base + (size_t)w * 64 + c] = f2bf(swishf(tile[c][w] + s1[b * 64 + c]));
    }
}

// ---------------------------------------------------------------------------
// conv20: g = swish(conv3x3x3(a0) + b20) * s2 ; 27 stages of K=64 (one tap)
// ---------------------------------------------------------------------------
__global__ __launch_bounds__(256) void conv20_wmma_kernel(
    const u16* __restrict__ a0, const u16* __restrict__ wT,
    const float* __restrict__ b20, const float* __restrict__ s2,
    u16* __restrict__ g)
{
    __shared__ __align__(16) u16 lA[2][128 * 64];
    __shared__ __align__(16) u16 lB[2][128 * 64];
    const int tid = threadIdx.x, wid = tid >> 5, lane = tid & 31;
    const int bid = blockIdx.x;
    const int b = bid >> 10, z = (bid >> 7) & 7, y = bid & 127;
    const int arow = wid * 16 + (lane & 15);

    auto stage = [&](int tap, int buf) {
        const int dz = tap / 9, dy = (tap / 3) % 3, dw = tap % 3;
        const size_t abase = (size_t)((b * DP + z + dz) * HP + (y + dy)) * WP * 64;
        const u16* wbase = wT + (size_t)tap * 8192;
        u16* dA = lA[buf]; u16* dB = lB[buf];
#pragma unroll
        for (int i = 0; i < 4; ++i) {
            int q = i * 256 + tid, row = q >> 3, oct = q & 7;
            cp16_lds(dA + q * 8, a0 + abase + (size_t)(row + dw) * 64 + oct * 8);
            cp16_lds(dB + q * 8, wbase + (size_t)row * 64 + oct * 8);
        }
    };

    v8f acc[8] = {};
    stage(0, 0);
    wait_stage();
    for (int s = 0; s < 27; ++s) {
        if (s < 26) stage(s + 1, (s + 1) & 1);
        compute_stage(lA[s & 1], lB[s & 1], arow, lane, acc);
        wait_stage();
    }

    const size_t gbase = ((size_t)((b * DP + z + 1) * HP + (y + 1)) * WP + 1) * 128;
#pragma unroll
    for (int nb = 0; nb < 8; ++nb) {
        const int n = nb * 16 + (lane & 15);
        const float bias = b20[n], sc = s2[b * 128 + n];
#pragma unroll
        for (int j = 0; j < 8; ++j) {
            const int p = wid * 16 + ((lane < 16) ? j : j + 8);
            g[gbase + (size_t)p * 128 + n] = f2bf(swishf(acc[nb][j] + bias) * sc);
        }
    }
}

// ---------------------------------------------------------------------------
// conv21 + residual: hs = swish(conv3x3x3(g) + conv1x1(x, res_w) + res_b)
// 54 main stages (27 taps x 2 halves) + 1 residual stage (K=64 over raw x)
// ---------------------------------------------------------------------------
__global__ __launch_bounds__(256) void conv21_wmma_kernel(
    const u16* __restrict__ g, const u16* __restrict__ wT,
    const float* __restrict__ x, const u16* __restrict__ resT,
    const float* __restrict__ res_b, u16* __restrict__ hs)
{
    __shared__ __align__(16) u16 lA[2][128 * 64];
    __shared__ __align__(16) u16 lB[2][128 * 64];
    const int tid = threadIdx.x, wid = tid >> 5, lane = tid & 31;
    const int bid = blockIdx.x;
    const int b = bid >> 10, z = (bid >> 7) & 7, y = bid & 127;
    const int arow = wid * 16 + (lane & 15);

    auto stage = [&](int s, int buf) {
        const int tap = s >> 1, c0 = (s & 1) << 6;
        const int dz = tap / 9, dy = (tap / 3) % 3, dw = tap % 3;
        const size_t abase = (size_t)((b * DP + z + dz) * HP + (y + dy)) * WP * 128;
        const u16* wbase = wT + (size_t)tap * 16384 + c0;
        u16* dA = lA[buf]; u16* dB = lB[buf];
#pragma unroll
        for (int i = 0; i < 4; ++i) {
            int q = i * 256 + tid, row = q >> 3, oct = q & 7;
            cp16_lds(dA + q * 8, g + abase + (size_t)(row + dw) * 128 + c0 + oct * 8);
            cp16_lds(dB + q * 8, wbase + (size_t)row * 128 + oct * 8);
        }
    };
    auto stage_res = [&](int buf) {        // residual 1x1 conv: A = bf16(raw x)
        u16* dA = lA[buf]; u16* dB = lB[buf];
        for (int i = 0; i < 32; ++i) {
            int q = i * 256 + tid, ci = q >> 7, w = q & 127;
            dA[w * 64 + ci] =
                f2bf(x[(((size_t)(b * 64 + ci) * 8 + z) * 128 + y) * 128 + w]);
        }
#pragma unroll
        for (int i = 0; i < 4; ++i) {
            int q = i * 256 + tid;
            cp16_lds(dB + q * 8, resT + q * 8);
        }
    };

    v8f acc[8] = {};
    stage(0, 0);
    wait_stage();
    for (int s = 0; s < 54; ++s) {
        if (s < 53) stage(s + 1, (s + 1) & 1);
        else        stage_res(0);          // (53+1)&1 == 0
        compute_stage(lA[s & 1], lB[s & 1], arow, lane, acc);
        wait_stage();
    }
    compute_stage(lA[0], lB[0], arow, lane, acc);   // residual stage

    const size_t hbase = ((size_t)((b * DP + z + 1) * HP + (y + 1)) * WP + 1) * 128;
#pragma unroll
    for (int nb = 0; nb < 8; ++nb) {
        const int n = nb * 16 + (lane & 15);
        const float bias = res_b[n];
#pragma unroll
        for (int j = 0; j < 8; ++j) {
            const int p = wid * 16 + ((lane < 16) ? j : j + 8);
            hs[hbase + (size_t)p * 128 + n] = f2bf(swishf(acc[nb][j] + bias));
        }
    }
}

// ---------------------------------------------------------------------------
// down: out = conv3x3x3(hs, stride (1,2,2), pad (1,0,0)) + down_b
// 54 stages of K=64; M=64 (x' 0..63, last column discarded at store)
// ---------------------------------------------------------------------------
__global__ __launch_bounds__(128) void down_wmma_kernel(
    const u16* __restrict__ hs, const u16* __restrict__ wT,
    const float* __restrict__ down_b, float* __restrict__ out)
{
    __shared__ __align__(16) u16 lA[2][64 * 64];
    __shared__ __align__(16) u16 lB[2][128 * 64];
    const int tid = threadIdx.x, wid = tid >> 5, lane = tid & 31;
    const int bid = blockIdx.x;
    const int b = bid / 504, rem = bid % 504, z = rem / 63, yq = rem % 63;
    const int arow = wid * 16 + (lane & 15);

    auto stage = [&](int s, int buf) {
        const int tap = s >> 1, c0 = (s & 1) << 6;
        const int dz = tap / 9, dy = (tap / 3) % 3, dw = tap % 3;
        const size_t rowbase = (size_t)((b * DP + z + dz) * HP + (2 * yq + dy + 1)) * WP;
        const u16* wbase = wT + (size_t)tap * 16384 + c0;
        u16* dA = lA[buf]; u16* dB = lB[buf];
#pragma unroll
        for (int i = 0; i < 4; ++i) {        // A: 64 rows, stride-2 gather
            int q = i * 128 + tid, row = q >> 3, oct = q & 7;
            int xp = 2 * row + dw + 1;
            cp16_lds(dA + q * 8, hs + (rowbase + xp) * 128 + c0 + oct * 8);
        }
#pragma unroll
        for (int i = 0; i < 8; ++i) {        // B: 128 rows
            int q = i * 128 + tid, n = q >> 3, oct = q & 7;
            cp16_lds(dB + q * 8, wbase + (size_t)n * 128 + oct * 8);
        }
    };

    v8f acc[8] = {};
    stage(0, 0);
    wait_stage();
    for (int s = 0; s < 54; ++s) {
        if (s < 53) stage(s + 1, (s + 1) & 1);
        compute_stage(lA[s & 1], lB[s & 1], arow, lane, acc);
        wait_stage();
    }

#pragma unroll
    for (int nb = 0; nb < 8; ++nb) {
        const int n = nb * 16 + (lane & 15);
        const float bias = down_b[n];
#pragma unroll
        for (int j = 0; j < 8; ++j) {
            const int p = wid * 16 + ((lane < 16) ? j : j + 8);
            if (p < 63)
                out[(((size_t)(b * 128 + n) * 8 + z) * 63 + yq) * 63 + p] = acc[nb][j] + bias;
        }
    }
}

// ---------------------------------------------------------------------------
extern "C" void kernel_launch(void* const* d_in, const int* in_sizes, int n_in,
                              void* d_out, int out_size, void* d_ws, size_t ws_size,
                              hipStream_t stream)
{
    const float* x       = (const float*)d_in[0];
    const float* embed   = (const float*)d_in[1];
    const float* context = (const float*)d_in[2];
    const float* w20     = (const float*)d_in[3];
    const float* b20     = (const float*)d_in[4];
    const float* w21     = (const float*)d_in[5];
    const float* d1w     = (const float*)d_in[6];
    const float* d1b     = (const float*)d_in[7];
    const float* d2w     = (const float*)d_in[8];
    const float* d2b     = (const float*)d_in[9];
    const float* resw    = (const float*)d_in[10];
    const float* resb    = (const float*)d_in[11];
    const float* dwnw    = (const float*)d_in[12];
    const float* dwnb    = (const float*)d_in[13];

    char* ws = (char*)d_ws;
    size_t o = 0;
    auto alloc = [&](size_t bytes) {
        void* p = ws + o;
        o = (o + bytes + 255) & ~(size_t)255;
        return p;
    };
    float* s1 = (float*)alloc(256 * 4);
    float* s2 = (float*)alloc(512 * 4);
    u16* wT20 = (u16*)alloc(221184 * 2);
    u16* wT21 = (u16*)alloc(442368 * 2);
    u16* wTdn = (u16*)alloc(442368 * 2);
    u16* resT = (u16*)alloc(8192 * 2);
    const size_t A0_ELEMS = (size_t)4 * DP * HP * WP * 64;
    const size_t G_ELEMS  = (size_t)4 * DP * HP * WP * 128;
    u16* a0 = (u16*)alloc(A0_ELEMS * 2);
    u16* g  = (u16*)alloc(G_ELEMS * 2);
    u16* hs = (u16*)alloc(G_ELEMS * 2);

    const long zero_u4 = (long)((A0_ELEMS + 2 * G_ELEMS) * 2 / 16);
    zero_fill_kernel<<<4096, 256, 0, stream>>>((uint4*)a0, zero_u4);

    modulate_kernel<<<3, 256, 0, stream>>>(embed, context, d1w, d1b, d2w, d2b, s1, s2);
    prep_weights_kernel<<<4352, 256, 0, stream>>>(w20, w21, dwnw, resw, wT20, wT21, wTdn, resT);
    prep_a0_kernel<<<4096, 256, 0, stream>>>(x, s1, a0);

    conv20_wmma_kernel<<<4096, 256, 0, stream>>>(a0, wT20, b20, s2, g);
    conv21_wmma_kernel<<<4096, 256, 0, stream>>>(g, wT21, x, resT, resb, hs);
    down_wmma_kernel<<<2016, 128, 0, stream>>>(hs, wTdn, dwnb, (float*)d_out);
}
